// PVCNNDiffusion_89094801589054
// MI455X (gfx1250) — matrix operand order repaired and checked by reading, not compiled, not run on hardware
//
#include <hip/hip_runtime.h>
#include <hip/hip_bf16.h>

#define BB 16
#define NP 65536
#define RV 32
#define CH 32
#define RH 34
#define KDIM 128          // padded K (27 taps * 4ch)
#define KREAL 108
#define NVOX 32768        // 32^3

typedef __attribute__((ext_vector_type(16))) __bf16 v16bf;
typedef __attribute__((ext_vector_type(8)))  __bf16 v8bf;
typedef __attribute__((ext_vector_type(4)))  __bf16 v4bf;
typedef __attribute__((ext_vector_type(8)))  float  v8f;
typedef __attribute__((ext_vector_type(4)))  float  v4f;

// ---- workspace layout (bytes) ----
#define OFF_SUM   0ull
#define SZ_SUM    ((unsigned long long)BB*NVOX*3*4)     // 6,291,456
#define OFF_CNT   (OFF_SUM + SZ_SUM)
#define SZ_CNT    ((unsigned long long)BB*NVOX*4)       // 2,097,152
#define OFF_HG    (OFF_CNT + SZ_CNT)                    // 8,388,608
#define SZ_HG     ((unsigned long long)BB*RH*RH*RH*4*2) // 5,030,912 (bf16 x4ch)
#define OFF_WT    (OFF_HG + SZ_HG)                      // 13,419,520
#define SZ_WT     ((unsigned long long)CH*KDIM*2)       // 8,192
#define OFF_CONV  (OFF_WT + SZ_WT)                      // 13,427,712 (1KB aligned)
#define ZERO_BYTES OFF_WT                               // zero sum+cnt+halo grid

// -------------------- zero scratch --------------------
__global__ void k_zero(uint4* __restrict__ p, int n) {
    int stride = gridDim.x * blockDim.x;
    for (int i = blockIdx.x * blockDim.x + threadIdx.x; i < n; i += stride) {
        uint4 z; z.x = 0u; z.y = 0u; z.z = 0u; z.w = 0u;
        p[i] = z;
    }
}

// -------------------- voxelize (scatter-add) --------------------
__global__ void k_vox(const float* __restrict__ coords, const float* __restrict__ feats,
                      float* __restrict__ sum, float* __restrict__ cnt) {
    int gid = blockIdx.x * blockDim.x + threadIdx.x;      // < BB*NP
    int b = gid >> 16;
    size_t p = (size_t)gid * 3;
    float cx = coords[p], cy = coords[p + 1], cz = coords[p + 2];
    float fx = feats[p],  fy = feats[p + 1],  fz = feats[p + 2];
    float nx = fminf(fmaxf(cx, 0.f), 1.f) * 31.f;
    float ny = fminf(fmaxf(cy, 0.f), 1.f) * 31.f;
    float nz = fminf(fmaxf(cz, 0.f), 1.f) * 31.f;
    int ix = min((int)nx, 31), iy = min((int)ny, 31), iz = min((int)nz, 31);
    int cell = (b << 15) + (ix << 10) + (iy << 5) + iz;
    atomicAdd(&sum[(size_t)cell * 3 + 0], fx);
    atomicAdd(&sum[(size_t)cell * 3 + 1], fy);
    atomicAdd(&sum[(size_t)cell * 3 + 2], fz);
    atomicAdd(&cnt[cell], 1.f);
}

// -------------------- finalize: average -> bf16 halo grid --------------------
__global__ void k_fin(const float* __restrict__ sum, const float* __restrict__ cnt,
                      __bf16* __restrict__ hgrid) {
    int gid = blockIdx.x * blockDim.x + threadIdx.x;      // < BB*NVOX
    float inv = 1.f / fmaxf(cnt[gid], 1.f);
    int b = gid >> 15, idx = gid & 32767;
    int x = idx >> 10, y = (idx >> 5) & 31, z = idx & 31;
    v4bf o;
    o[0] = (__bf16)(sum[(size_t)gid * 3 + 0] * inv);
    o[1] = (__bf16)(sum[(size_t)gid * 3 + 1] * inv);
    o[2] = (__bf16)(sum[(size_t)gid * 3 + 2] * inv);
    o[3] = (__bf16)0.f;
    size_t off = ((size_t)((b * RH + x + 1) * RH + (y + 1)) * RH + (z + 1)) * 4;
    *(v4bf*)(hgrid + off) = o;
}

// -------------------- weight repack: OIDHW f32 -> Wt[co][k] bf16 --------------------
__global__ void k_prepw(const float* __restrict__ Wc, __bf16* __restrict__ Wt) {
    int i = blockIdx.x * blockDim.x + threadIdx.x;        // < CH*KDIM
    if (i >= CH * KDIM) return;
    int co = i >> 7, k = i & 127;
    float v = 0.f;
    if (k < KREAL) {
        int ci = k & 3, tap = k >> 2;                     // tap = dx*9+dy*3+dz
        if (ci < 3) v = Wc[(co * 3 + ci) * 27 + tap];
    }
    Wt[i] = (__bf16)v;
}

// -------------------- conv 3x3x3 as im2col GEMM via WMMA bf16 --------------------
__global__ void __launch_bounds__(256) k_conv(const __bf16* __restrict__ hgrid,
                                              const __bf16* __restrict__ Wt,
                                              const float* __restrict__ bconv,
                                              float* __restrict__ convout) {
    __shared__ __align__(16) __bf16 ldsA[128 * KDIM];     // 32 KB: 128 voxel rows x K=128
    __shared__ __align__(16) __bf16 ldsW[CH * KDIM];      // 8 KB:  Wt transposed [co][k]
    int tid = threadIdx.x;
    int batch = blockIdx.x >> 8;
    int vbase = (blockIdx.x & 255) * 128;

    // stage im2col A-panel asynchronously: 128 voxels x 27 taps, 8B (4 bf16 ch) per
    // tap, memory -> LDS directly via GLOBAL_LOAD_ASYNC_TO_LDS_B64 (ASYNCcnt path,
    // no VGPR landing registers). GVS mode: uniform batch base in SGPRs + 32-bit
    // per-lane byte offset in a VGPR; LDS address = low 32 bits of generic pointer.
    const __bf16* hb = hgrid + (size_t)batch * RH * RH * RH * 4;
    for (int i = tid; i < 128 * 27; i += 256) {
        int v = i / 27, tap = i - v * 27;
        int vb = vbase + v;
        int z = vb & 31, y = (vb >> 5) & 31, x = (vb >> 10) & 31;
        int dz = tap % 3, dy = (tap / 3) % 3, dx = tap / 9;
        unsigned goff = (unsigned)((((x + dx) * RH + (y + dy)) * RH + (z + dz)) * 8);
        unsigned laddr = (unsigned)(size_t)&ldsA[v * KDIM + tap * 4];
        asm volatile("global_load_async_to_lds_b64 %0, %1, %2"
                     :: "v"(laddr), "v"(goff), "s"(hb)
                     : "memory");
    }
    // stage weights (B-panel) through VGPRs while async transfers are in flight
    {
        const uint4* src = (const uint4*)Wt;
        uint4* dst = (uint4*)ldsW;
        for (int i = tid; i < (CH * KDIM * 2) / 16; i += 256) dst[i] = src[i];
    }
    // zero K padding [108,128)
    for (int i = tid; i < 128 * 5; i += 256) {
        int row = i / 5, j = i - row * 5;
        *(unsigned long long*)&ldsA[row * KDIM + KREAL + j * 4] = 0ull;
    }
    asm volatile("s_wait_asynccnt 0" ::: "memory");       // async LDS writes visible
    __syncthreads();

    int lane = tid & 31;
    int wave = tid >> 5;                                  // 8 wave32s
    int mrow = wave * 16 + (lane & 15);
    int selA = (lane >> 4) * 16;                          // A: hi lanes take K 16..31
    int selB = (lane >> 4) * 16;                          // B: hi lanes take K 16..31
    v8f acc0 = {}; v8f acc1 = {};
#pragma unroll
    for (int ks = 0; ks < 4; ks++) {
        const __bf16* ap = &ldsA[mrow * KDIM + ks * 32];
        v8bf aLo = *(const v8bf*)(ap + selA);
        v8bf aHi = *(const v8bf*)(ap + 8 + selA);
        v16bf a;
#pragma unroll
        for (int e = 0; e < 8; e++) { a[e] = aLo[e]; a[8 + e] = aHi[e]; }

        const __bf16* wp0 = &ldsW[(lane & 15) * KDIM + ks * 32 + selB];
        v8bf b0l = *(const v8bf*)wp0;
        v8bf b0h = *(const v8bf*)(wp0 + 8);
        v16bf b0;
#pragma unroll
        for (int e = 0; e < 8; e++) { b0[e] = b0l[e]; b0[8 + e] = b0h[e]; }

        const __bf16* wp1 = &ldsW[(16 + (lane & 15)) * KDIM + ks * 32 + selB];
        v8bf b1l = *(const v8bf*)wp1;
        v8bf b1h = *(const v8bf*)(wp1 + 8);
        v16bf b1;
#pragma unroll
        for (int e = 0; e < 8; e++) { b1[e] = b1l[e]; b1[8 + e] = b1h[e]; }

        acc0 = __builtin_amdgcn_wmma_f32_16x16x32_bf16(false, a, false, b0, (short)0, acc0, false, false);
        acc1 = __builtin_amdgcn_wmma_f32_16x16x32_bf16(false, a, false, b1, (short)0, acc1, false, false);
    }

    // bias + ReLU + store channel-last (C layout: VGPR r -> M = r (+8 for hi lanes), N = lane%16)
    int n = lane & 15;
    int mofs = (lane >> 4) * 8;
    float bn0 = bconv[n], bn1 = bconv[n + 16];
    float* ob = convout + ((size_t)batch * NVOX + vbase + wave * 16) * CH;
#pragma unroll
    for (int r = 0; r < 8; r++) {
        float* orow = ob + (size_t)(r + mofs) * CH;
        orow[n]      = fmaxf(acc0[r] + bn0, 0.f);
        orow[n + 16] = fmaxf(acc1[r] + bn1, 0.f);
    }
}

// -------------------- fused devoxelize + point MLP + time embed --------------------
__global__ void __launch_bounds__(256) k_fuse(const float* __restrict__ conv,
                                              const float* __restrict__ coords,
                                              const float* __restrict__ feats,
                                              const float* __restrict__ t,
                                              const float* __restrict__ Wp,
                                              const float* __restrict__ bp,
                                              const float* __restrict__ Wtm,
                                              const float* __restrict__ btm,
                                              float* __restrict__ out) {
    int gid = blockIdx.x * blockDim.x + threadIdx.x;      // < BB*NP
    int b = gid >> 16;
    size_t p = (size_t)gid * 3;
    float cx = coords[p], cy = coords[p + 1], cz = coords[p + 2];
    float fx = feats[p],  fy = feats[p + 1],  fz = feats[p + 2];
    float nx = fminf(fmaxf(cx, 0.f), 1.f) * 31.f;
    float ny = fminf(fmaxf(cy, 0.f), 1.f) * 31.f;
    float nz = fminf(fmaxf(cz, 0.f), 1.f) * 31.f;
    int lx = min((int)nx, 31), ly = min((int)ny, 31), lz = min((int)nz, 31);
    int hx = min(lx + 1, 31), hy = min(ly + 1, 31), hz = min(lz + 1, 31);
    float frx = nx - (float)lx, fry = ny - (float)ly, frz = nz - (float)lz;

    float acc[CH];
    float tb = t[b];
#pragma unroll
    for (int c = 0; c < CH; c++) {
        float pm = fx * Wp[c] + fy * Wp[CH + c] + fz * Wp[2 * CH + c] + bp[c];
        acc[c] = fmaxf(pm, 0.f) + tb * Wtm[c] + btm[c];
    }
    const float* cb = conv + (size_t)b * NVOX * CH;
#pragma unroll
    for (int d = 0; d < 8; d++) {
        int ix = (d & 4) ? hx : lx;
        int iy = (d & 2) ? hy : ly;
        int iz = (d & 1) ? hz : lz;
        float w = ((d & 4) ? frx : 1.f - frx) *
                  ((d & 2) ? fry : 1.f - fry) *
                  ((d & 1) ? frz : 1.f - frz);
        const v4f* src = (const v4f*)(cb + (size_t)(((ix << 5) + iy) * 32 + iz) * CH);
#pragma unroll
        for (int q = 0; q < 8; q++) {
            v4f v = src[q];
            acc[q * 4 + 0] += w * v.x; acc[q * 4 + 1] += w * v.y;
            acc[q * 4 + 2] += w * v.z; acc[q * 4 + 3] += w * v.w;
        }
    }
    v4f* op = (v4f*)(out + (size_t)gid * CH);
#pragma unroll
    for (int q = 0; q < 8; q++) {
        v4f v; v.x = acc[4 * q]; v.y = acc[4 * q + 1]; v.z = acc[4 * q + 2]; v.w = acc[4 * q + 3];
        __builtin_nontemporal_store(v, op + q);
    }
}

extern "C" void kernel_launch(void* const* d_in, const int* in_sizes, int n_in,
                              void* d_out, int out_size, void* d_ws, size_t ws_size,
                              hipStream_t stream) {
    const float* features = (const float*)d_in[0];
    const float* coords   = (const float*)d_in[1];
    const float* t        = (const float*)d_in[2];
    const float* W_point  = (const float*)d_in[3];
    const float* b_point  = (const float*)d_in[4];
    const float* W_conv   = (const float*)d_in[5];
    const float* b_conv   = (const float*)d_in[6];
    const float* W_time   = (const float*)d_in[7];
    const float* b_time   = (const float*)d_in[8];
    float* out = (float*)d_out;
    char* ws = (char*)d_ws;

    float*  vox_sum = (float*)(ws + OFF_SUM);
    float*  vox_cnt = (float*)(ws + OFF_CNT);
    __bf16* hgrid   = (__bf16*)(ws + OFF_HG);
    __bf16* Wt      = (__bf16*)(ws + OFF_WT);
    float*  convout = (float*)(ws + OFF_CONV);

    k_zero<<<1024, 256, 0, stream>>>((uint4*)ws, (int)(ZERO_BYTES / 16));
    k_vox<<<(BB * NP) / 256, 256, 0, stream>>>(coords, features, vox_sum, vox_cnt);
    k_fin<<<(BB * NVOX) / 256, 256, 0, stream>>>(vox_sum, vox_cnt, hgrid);
    k_prepw<<<(CH * KDIM) / 256, 256, 0, stream>>>(W_conv, Wt);
    k_conv<<<BB * (NVOX / 128), 256, 0, stream>>>(hgrid, Wt, b_conv, convout);
    k_fuse<<<(BB * NP) / 256, 256, 0, stream>>>(convout, coords, features, t,
                                                W_point, b_point, W_time, b_time, out);
}
